// VanillaRNN_81355270521389
// MI455X (gfx1250) — matrix operand order repaired and checked by reading, not compiled, not run on hardware
//
#include <hip/hip_runtime.h>
#include <stdint.h>

#define VOCAB 32000
#define EMB   1024
#define HID   1024
#define BATCH 16
#define SEQT  256

// ---- CDNA5 WMMA types (wave32, 16x16x32 bf16 -> f32) ----
typedef __attribute__((ext_vector_type(16))) __bf16 v16bf;
typedef __attribute__((ext_vector_type(8)))  __bf16 v8bf;
typedef __attribute__((ext_vector_type(8)))  float  v8f;
typedef __attribute__((ext_vector_type(4)))  int    v4i;

union ABf16 { v16bf v; v8bf h[2]; };

#if __has_builtin(__builtin_amdgcn_global_load_async_to_lds_b128)
#define HAVE_ASYNC_COPY 1
typedef __attribute__((address_space(1))) v4i gv4i; // global int4
typedef __attribute__((address_space(3))) v4i lv4i; // LDS int4
#endif

__device__ __forceinline__ unsigned short f32_to_bf16(float f) {
  union { float f; unsigned int u; } x; x.f = f;
  unsigned int r = 0x7FFFu + ((x.u >> 16) & 1u);   // round-to-nearest-even
  return (unsigned short)((x.u + r) >> 16);
}

__device__ __forceinline__ uint2 pack4_bf16(float4 f) {
  uint2 r;
  r.x = (unsigned)f32_to_bf16(f.x) | ((unsigned)f32_to_bf16(f.y) << 16);
  r.y = (unsigned)f32_to_bf16(f.z) | ((unsigned)f32_to_bf16(f.w) << 16);
  return r;
}

__device__ __forceinline__ v8f wmma_bf16(const ABf16& a, const ABf16& b, v8f c) {
  // 8 args: (neg_a, A, neg_b, B, c_mod, C, reuse_a, reuse_b)
  return __builtin_amdgcn_wmma_f32_16x16x32_bf16(false, a.v, false, b.v,
                                                 (short)0, c, false, false);
}

// ================= prep kernels =================

__global__ void k_f32_to_bf16x4(const float* __restrict__ src,
                                unsigned short* __restrict__ dst, size_t n4) {
  size_t i = (size_t)blockIdx.x * blockDim.x + threadIdx.x;
  if (i < n4) {
    float4 f = ((const float4*)src)[i];
    ((uint2*)dst)[i] = pack4_bf16(f);
  }
}

// transpose 1024x1024 f32 -> bf16 (dst[n*HID+k] = src[k*HID+n]); blockIdx.z picks matrix
__global__ void k_transpose_to_bf16(const float* __restrict__ s0, const float* __restrict__ s1,
                                    const float* __restrict__ s2, const float* __restrict__ s3,
                                    unsigned short* __restrict__ d0, unsigned short* __restrict__ d1,
                                    unsigned short* __restrict__ d2, unsigned short* __restrict__ d3) {
  const float* src; unsigned short* dst;
  switch (blockIdx.z) {
    case 0:  src = s0; dst = d0; break;
    case 1:  src = s1; dst = d1; break;
    case 2:  src = s2; dst = d2; break;
    default: src = s3; dst = d3; break;
  }
  __shared__ float tile[32][33];
  int bx = blockIdx.x * 32, by = blockIdx.y * 32;
  int tx = threadIdx.x, ty = threadIdx.y;          // block (32,8)
  #pragma unroll
  for (int i = 0; i < 32; i += 8)
    tile[ty + i][tx] = src[(size_t)(by + ty + i) * HID + bx + tx];
  __syncthreads();
  #pragma unroll
  for (int i = 0; i < 32; i += 8)
    dst[(size_t)(bx + ty + i) * HID + by + tx] = f32_to_bf16(tile[tx][ty + i]);
}

// gather embeddings time-major: emb_seq[(t*B+b)*EMB + k] = bf16(emb_W[x[b,t]][k])
__global__ void k_embed(const int* __restrict__ x, const float* __restrict__ embW,
                        unsigned short* __restrict__ emb_seq) {
  int tb = blockIdx.x;                 // 0..T*B-1 : t = tb>>4, b = tb&15
  int t = tb >> 4, b = tb & 15;
  int vi = x[b * SEQT + t];
  const float4* src = (const float4*)(embW + (size_t)vi * EMB);
  uint2* dst = (uint2*)(emb_seq + (size_t)tb * EMB);
  float4 f = src[threadIdx.x];         // 256 threads * 4 floats = 1024
  dst[threadIdx.x] = pack4_bf16(f);
}

// init bf16 h-state (buffer 0) from f32 hidden (2,16,1024)
__global__ void k_init_h(const float* __restrict__ hidden,
                         unsigned short* __restrict__ h0buf,
                         unsigned short* __restrict__ h1buf) {
  int i = blockIdx.x * blockDim.x + threadIdx.x;   // 0..16*1024-1
  h0buf[i] = f32_to_bf16(hidden[i]);
  h1buf[i] = f32_to_bf16(hidden[BATCH * HID + i]);
}

// ================= device-wide generation barrier =================
__device__ __forceinline__ void grid_barrier(int* count, int* gen, int nblocks) {
  __syncthreads();
  if (threadIdx.x == 0) {
    __threadfence();
    int g = __hip_atomic_load(gen, __ATOMIC_ACQUIRE, __HIP_MEMORY_SCOPE_AGENT);
    int a = __hip_atomic_fetch_add(count, 1, __ATOMIC_ACQ_REL, __HIP_MEMORY_SCOPE_AGENT);
    if (a == nblocks - 1) {
      __hip_atomic_store(count, 0, __ATOMIC_RELAXED, __HIP_MEMORY_SCOPE_AGENT);
      __hip_atomic_fetch_add(gen, 1, __ATOMIC_ACQ_REL, __HIP_MEMORY_SCOPE_AGENT);
    } else {
      while (__hip_atomic_load(gen, __ATOMIC_ACQUIRE, __HIP_MEMORY_SCOPE_AGENT) == g)
        __builtin_amdgcn_s_sleep(2);
    }
  }
  __syncthreads();
}

// ================= sequential RNN scan (persistent) =================
// 16 blocks x 128 threads = 64 waves; wave wid owns N-columns [wid*16, wid*16+16)
__global__ __launch_bounds__(128) void k_rnn_scan(
    const unsigned short* __restrict__ emb_seq,
    const unsigned short* __restrict__ Wx0T, const unsigned short* __restrict__ Wh0T,
    const unsigned short* __restrict__ Wx1T, const unsigned short* __restrict__ Wh1T,
    const float* __restrict__ b0, const float* __restrict__ b1,
    unsigned short* __restrict__ h0buf, unsigned short* __restrict__ h1buf,
    unsigned short* __restrict__ h1_all, float* __restrict__ out_hidden,
    int* bar_count, int* bar_gen, int nblocks) {
  const int wid  = (blockIdx.x * blockDim.x + threadIdx.x) >> 5;
  const int lane = threadIdx.x & 31;
  const int nn = (wid << 4) + (lane & 15);         // output column
  const int m  = lane & 15;                        // A row owned by this lane
  const int mb = (lane >> 4) << 3;                 // C rows: mb..mb+7
  const int ka = (lane < 16) ? 0 : 8;              // A K-offset (ISA layout)
  const int kb = (lane < 16) ? 0 : 8;              // B K-offset (ISA layout)
  const float bias0 = b0[nn];
  const float bias1 = b1[nn];
  const unsigned short* wx0r = Wx0T + (size_t)nn * HID;
  const unsigned short* wh0r = Wh0T + (size_t)nn * HID;
  const unsigned short* wx1r = Wx1T + (size_t)nn * HID;
  const unsigned short* wh1r = Wh1T + (size_t)nn * HID;
  const v8f vzero = {};

  for (int t = 0; t < SEQT; ++t) {
    const int cur = t & 1, nxt = cur ^ 1;
    const unsigned short* eA  = emb_seq + ((size_t)t * BATCH + m) * EMB;
    const unsigned short* h0A = h0buf + cur * BATCH * HID + m * HID;
    const unsigned short* h1A = h1buf + cur * BATCH * HID + m * HID;
    unsigned short* h0D = h0buf + nxt * BATCH * HID;
    unsigned short* h1D = h1buf + nxt * BATCH * HID;

    // ---- layer 0: h0n = tanh(e@Wx0 + h0@Wh0 + b0) ----
    v8f acc = vzero;
    #pragma unroll 4
    for (int k0 = 0; k0 < HID; k0 += 32) {
      ABf16 a, b;
      a.h[0] = *(const v8bf*)(eA + k0 + ka);
      a.h[1] = *(const v8bf*)(eA + k0 + 16 + ka);
      b.h[0] = *(const v8bf*)(wx0r + k0 + kb);
      b.h[1] = *(const v8bf*)(wx0r + k0 + 16 + kb);
      acc = wmma_bf16(a, b, acc);
      a.h[0] = *(const v8bf*)(h0A + k0 + ka);
      a.h[1] = *(const v8bf*)(h0A + k0 + 16 + ka);
      b.h[0] = *(const v8bf*)(wh0r + k0 + kb);
      b.h[1] = *(const v8bf*)(wh0r + k0 + 16 + kb);
      acc = wmma_bf16(a, b, acc);
    }
    #pragma unroll
    for (int r = 0; r < 8; ++r) {
      float v = tanhf(acc[r] + bias0);
      h0D[(mb + r) * HID + nn] = f32_to_bf16(v);
      if (t == SEQT - 1) out_hidden[(mb + r) * HID + nn] = v;
    }
    grid_barrier(bar_count, bar_gen, nblocks);

    // ---- layer 1: h1n = tanh(h0n@Wx1 + h1@Wh1 + b1) ----
    const unsigned short* h0nA = h0D + m * HID;
    acc = vzero;
    #pragma unroll 4
    for (int k0 = 0; k0 < HID; k0 += 32) {
      ABf16 a, b;
      a.h[0] = *(const v8bf*)(h0nA + k0 + ka);
      a.h[1] = *(const v8bf*)(h0nA + k0 + 16 + ka);
      b.h[0] = *(const v8bf*)(wx1r + k0 + kb);
      b.h[1] = *(const v8bf*)(wx1r + k0 + 16 + kb);
      acc = wmma_bf16(a, b, acc);
      a.h[0] = *(const v8bf*)(h1A + k0 + ka);
      a.h[1] = *(const v8bf*)(h1A + k0 + 16 + ka);
      b.h[0] = *(const v8bf*)(wh1r + k0 + kb);
      b.h[1] = *(const v8bf*)(wh1r + k0 + 16 + kb);
      acc = wmma_bf16(a, b, acc);
    }
    #pragma unroll
    for (int r = 0; r < 8; ++r) {
      float v = tanhf(acc[r] + bias1);
      unsigned short us = f32_to_bf16(v);
      h1D[(mb + r) * HID + nn] = us;
      h1_all[((size_t)t * BATCH + mb + r) * HID + nn] = us;
      if (t == SEQT - 1) out_hidden[BATCH * HID + (mb + r) * HID + nn] = v;
    }
    grid_barrier(bar_count, bar_gen, nblocks);
  }
}

// ================= tied output projection =================
// logits(4096x32000) = h1_all(4096x1024) @ embW_bf16^T + out_b
// Register-blocked M4xN2: block stages 4 t-tiles (M=64) in 128KB LDS; each of 8
// waves owns 2 N-tiles whose B registers are reused across all 4 M-tiles
// (4x less L2 B-traffic). grid (T/4, VOCAB/16/16) = (64,125) -- no tails, so
// EXEC is all-ones for every WMMA.
__global__ __launch_bounds__(256) void k_proj(
    const unsigned short* __restrict__ h1_all,
    const unsigned short* __restrict__ embW,
    const float* __restrict__ out_b,
    float* __restrict__ out) {
  __shared__ __attribute__((aligned(16))) unsigned short Atile[4 * BATCH * HID]; // 128 KB
  const int t0 = blockIdx.x * 4;
  {
    const uint4* s = (const uint4*)(h1_all + (size_t)t0 * BATCH * HID);
    uint4* d = (uint4*)Atile;
#ifdef HAVE_ASYNC_COPY
    // CDNA5 async global->LDS DMA: no VGPR round trip, tracked by ASYNCcnt.
    #pragma unroll
    for (int i = 0; i < 32; ++i) {
      int idx = threadIdx.x + i * 256;               // 8192 uint4 total
      __builtin_amdgcn_global_load_async_to_lds_b128(
          (gv4i*)(uintptr_t)(s + idx),
          (lv4i*)(unsigned int)(uintptr_t)(d + idx),
          0, 0);
    }
#if __has_builtin(__builtin_amdgcn_s_wait_asynccnt)
    __builtin_amdgcn_s_wait_asynccnt(0);
#else
    asm volatile("s_wait_asynccnt 0" ::: "memory");
#endif
#else
    for (int i = threadIdx.x; i < 4 * BATCH * HID / 8; i += 256) d[i] = s[i];
#endif
  }
  __syncthreads();

  const int wv = threadIdx.x >> 5, lane = threadIdx.x & 31;
  const int m  = lane & 15;
  const int mb = (lane >> 4) << 3;
  const int ka = (lane < 16) ? 0 : 8;
  const int kb = (lane < 16) ? 0 : 8;
  const int nt0 = blockIdx.y * 16 + wv * 2;          // 2 consecutive N-tiles/wave
  const int nn0 = nt0 * 16 + m;
  const int nn1 = nn0 + 16;
  const unsigned short* brow0 = embW + (size_t)nn0 * EMB; // embW row == B column
  const unsigned short* brow1 = embW + (size_t)nn1 * EMB;

  v8f acc[4][2];
  const v8f vz = {};
  #pragma unroll
  for (int mi = 0; mi < 4; ++mi) { acc[mi][0] = vz; acc[mi][1] = vz; }

  #pragma unroll 2
  for (int k0 = 0; k0 < EMB; k0 += 32) {
    ABf16 b0v, b1v;
    b0v.h[0] = *(const v8bf*)(brow0 + k0 + kb);
    b0v.h[1] = *(const v8bf*)(brow0 + k0 + 16 + kb);
    b1v.h[0] = *(const v8bf*)(brow1 + k0 + kb);
    b1v.h[1] = *(const v8bf*)(brow1 + k0 + 16 + kb);
    __builtin_prefetch(brow0 + k0 + 512, 0, 1);      // global_prefetch_b8
    __builtin_prefetch(brow1 + k0 + 512, 0, 1);
    #pragma unroll
    for (int mi = 0; mi < 4; ++mi) {
      const unsigned short* arow = Atile + (mi * BATCH + m) * HID;
      ABf16 a;
      a.h[0] = *(const v8bf*)(arow + k0 + ka);
      a.h[1] = *(const v8bf*)(arow + k0 + 16 + ka);
      acc[mi][0] = wmma_bf16(a, b0v, acc[mi][0]);
      acc[mi][1] = wmma_bf16(a, b1v, acc[mi][1]);
    }
  }

  const float bias0 = out_b[nn0];
  const float bias1 = out_b[nn1];
  #pragma unroll
  for (int mi = 0; mi < 4; ++mi) {
    const int t = t0 + mi;
    #pragma unroll
    for (int r = 0; r < 8; ++r) {
      size_t rowb = ((size_t)(mb + r) * SEQT + t) * (size_t)VOCAB; // out[b][t][v]
      out[rowb + nn0] = acc[mi][0][r] + bias0;
      out[rowb + nn1] = acc[mi][1][r] + bias1;
    }
  }
}

// ================= host glue =================
extern "C" void kernel_launch(void* const* d_in, const int* in_sizes, int n_in,
                              void* d_out, int out_size, void* d_ws, size_t ws_size,
                              hipStream_t stream) {
  (void)in_sizes; (void)n_in; (void)out_size; (void)ws_size;
  const int*   x      = (const int*)d_in[0];
  const float* hidden = (const float*)d_in[1];
  const float* embW   = (const float*)d_in[2];
  const float* Wx0    = (const float*)d_in[3];
  const float* Wh0    = (const float*)d_in[4];
  const float* b0     = (const float*)d_in[5];
  const float* Wx1    = (const float*)d_in[6];
  const float* Wh1    = (const float*)d_in[7];
  const float* b1     = (const float*)d_in[8];
  const float* out_b  = (const float*)d_in[9];
  float* out = (float*)d_out;
  float* out_hidden = out + (size_t)BATCH * SEQT * VOCAB;

  unsigned char* wsb = (unsigned char*)d_ws;
  size_t off = 0;
  auto take = [&](size_t bytes) -> void* {
    void* p = (void*)(wsb + off);
    off += (bytes + 255) & ~(size_t)255;
    return p;
  };
  unsigned short* embW_bf = (unsigned short*)take((size_t)VOCAB * EMB * 2);
  unsigned short* Wx0T    = (unsigned short*)take((size_t)HID * HID * 2);
  unsigned short* Wh0T    = (unsigned short*)take((size_t)HID * HID * 2);
  unsigned short* Wx1T    = (unsigned short*)take((size_t)HID * HID * 2);
  unsigned short* Wh1T    = (unsigned short*)take((size_t)HID * HID * 2);
  unsigned short* emb_seq = (unsigned short*)take((size_t)SEQT * BATCH * EMB * 2);
  unsigned short* h0buf   = (unsigned short*)take((size_t)2 * BATCH * HID * 2);
  unsigned short* h1buf   = (unsigned short*)take((size_t)2 * BATCH * HID * 2);
  unsigned short* h1_all  = (unsigned short*)take((size_t)SEQT * BATCH * HID * 2);
  int* bar = (int*)take(256);

  (void)hipMemsetAsync(bar, 0, 256, stream);

  // emb_W f32 -> bf16 (also serves as the transposed projection matrix)
  {
    size_t n4 = (size_t)VOCAB * EMB / 4;
    k_f32_to_bf16x4<<<(unsigned)((n4 + 255) / 256), 256, 0, stream>>>(embW, embW_bf, n4);
  }
  // recurrent weights: transpose + bf16
  k_transpose_to_bf16<<<dim3(HID / 32, HID / 32, 4), dim3(32, 8), 0, stream>>>(
      Wx0, Wh0, Wx1, Wh1, Wx0T, Wh0T, Wx1T, Wh1T);
  // embedding gather, time-major bf16
  k_embed<<<SEQT * BATCH, 256, 0, stream>>>(x, embW, emb_seq);
  // initial hidden state
  k_init_h<<<(BATCH * HID) / 256, 256, 0, stream>>>(hidden, h0buf, h1buf);
  // sequential scan: 16 blocks x 4 waves, device-wide barriers
  k_rnn_scan<<<16, 128, 0, stream>>>(emb_seq, Wx0T, Wh0T, Wx1T, Wh1T, b0, b1,
                                     h0buf, h1buf, h1_all, out_hidden,
                                     bar, bar + 1, 16);
  // batched tied projection (268 GFLOP): M4xN2 register blocking
  k_proj<<<dim3(SEQT / 4, VOCAB / 16 / 16), 256, 0, stream>>>(h1_all, embW_bf, out_b, out);
}
